// GraphAttentionLayer_8246337208569
// MI455X (gfx1250) — matrix operand order repaired and checked by reading, not compiled, not run on hardware
//
#include <hip/hip_runtime.h>

typedef __attribute__((ext_vector_type(16))) __bf16 v16bf;
typedef __attribute__((ext_vector_type(8)))  __bf16 v8bf;
typedef __attribute__((ext_vector_type(8)))  float  v8f;

#define ALPHA   0.2f
#define NEG_BIG -1.0e12f

#define BATCH 8
#define NNODE 2048
#define CDIM  256
#define NROW  (BATCH * NNODE)   // 16384
#define NWORD (NNODE / 32)      // 64 mask words per row

__device__ __forceinline__ __bf16 f2bf(float f) {
    union { float f; unsigned u; } a; a.f = f;
    unsigned r = a.u + 0x7FFFu + ((a.u >> 16) & 1u);   // round-to-nearest-even
    unsigned short h = (unsigned short)(r >> 16);
    return __builtin_bit_cast(__bf16, h);
}

// gfx1250 async global->LDS copy (16B per lane). Per ISA 10.2, the low 32 bits
// of a flat shared-aperture pointer are the LDS byte offset (VDST operand).
__device__ __forceinline__ void async_copy16(void* ldsp, const void* gp) {
    unsigned loff = (unsigned)(unsigned long long)ldsp;
    unsigned long long ga = (unsigned long long)gp;
    asm volatile("global_load_async_to_lds_b128 %0, %1, off"
                 :: "v"(loff), "v"(ga) : "memory");
}
__device__ __forceinline__ void wait_async0() {
    asm volatile("s_wait_asynccnt 0" ::: "memory");
}

// ---------------------------------------------------------------- converts
__global__ void cvt_x_kernel(const float* __restrict__ x,
                             __bf16* __restrict__ xb, int n) {
    int i = blockIdx.x * blockDim.x + threadIdx.x;
    if (i < n) xb[i] = f2bf(x[i]);
}

// wt layout: [3][cout=256][cin=256]  (transposed so B-fragments are contiguous in K)
__global__ void cvt_w_kernel(const float* __restrict__ w1,
                             const float* __restrict__ w2,
                             const float* __restrict__ w3,
                             __bf16* __restrict__ wt) {
    int i = blockIdx.x * blockDim.x + threadIdx.x;      // 0 .. 3*65536-1
    int z = i >> 16;
    int r = i & 65535;
    int cin = r >> 8, cout = r & 255;
    const float* w = (z == 0) ? w1 : (z == 1) ? w2 : w3;
    wt[z * 65536 + cout * 256 + cin] = f2bf(w[cin * 256 + cout]);
}

// pack adj rows into bitmasks: mb[row][w] bit k = adj[row][w*32+k] > 0
__global__ void pack_adj_kernel(const int* __restrict__ adj,
                                unsigned* __restrict__ mb) {
    int i = blockIdx.x * blockDim.x + threadIdx.x;      // 0 .. 2048*64-1
    if (i >= NNODE * NWORD) return;
    int row = i >> 6, word = i & (NWORD - 1);
    const int* p = adj + row * NNODE + word * 32;
    unsigned m = 0;
#pragma unroll
    for (int t = 0; t < 32; ++t) m |= (p[t] > 0 ? 1u : 0u) << t;
    mb[i] = m;
}

// ---------------------------------------------------------------- projections
// H = X @ W for W1,W2,W3.  Block = 8 waves, covers 16 rows x 256 cols.
// z==2 writes H3 transposed per-batch: h3t[b][c][n].
__global__ __launch_bounds__(256) void gemm_proj_kernel(
    const __bf16* __restrict__ xb,   // [16384][256]
    const __bf16* __restrict__ wt,   // [3][256][256] (cout-major)
    __bf16* __restrict__ h1,
    __bf16* __restrict__ h2,
    __bf16* __restrict__ h3t)        // [8][256][2048]
{
    const int z     = blockIdx.z;
    const int r0    = blockIdx.x * 16;
    const int wave  = threadIdx.x >> 5;
    const int lane  = threadIdx.x & 31;
    const int lhalf = lane >> 4;
    const int l16   = lane & 15;
    const int col0  = wave * 32;

    const __bf16* w    = wt + z * 65536;
    const __bf16* arow = xb + (r0 + l16) * CDIM;

    v8f acc0 = {}; v8f acc1 = {};
#pragma unroll
    for (int kc = 0; kc < 8; ++kc) {
        const int cb = kc * 32;
        v8bf alo = *(const v8bf*)(arow + cb + lhalf * 16);
        v8bf ahi = *(const v8bf*)(arow + cb + lhalf * 16 + 8);
        v16bf a  = __builtin_shufflevector(alo, ahi,
                       0,1,2,3,4,5,6,7,8,9,10,11,12,13,14,15);
        v16bf b0 = *(const v16bf*)(w + (col0      + l16) * CDIM + cb + lhalf * 16);
        v16bf b1 = *(const v16bf*)(w + (col0 + 16 + l16) * CDIM + cb + lhalf * 16);
        acc0 = __builtin_amdgcn_wmma_f32_16x16x32_bf16(false, a, false, b0,
                                                       (short)0, acc0, false, false);
        acc1 = __builtin_amdgcn_wmma_f32_16x16x32_bf16(false, a, false, b1,
                                                       (short)0, acc1, false, false);
    }

    if (z == 2) {
#pragma unroll
        for (int v = 0; v < 8; ++v) {
            int r = r0 + v + 8 * lhalf;
            int b = r >> 11, n = r & (NNODE - 1);
            h3t[b * (CDIM * NNODE) + (col0      + l16) * NNODE + n] = f2bf(acc0[v]);
            h3t[b * (CDIM * NNODE) + (col0 + 16 + l16) * NNODE + n] = f2bf(acc1[v]);
        }
    } else {
        __bf16* h = (z == 0) ? h1 : h2;
#pragma unroll
        for (int v = 0; v < 8; ++v) {
            int r = r0 + v + 8 * lhalf;
            h[r * CDIM + col0      + l16] = f2bf(acc0[v]);
            h[r * CDIM + col0 + 16 + l16] = f2bf(acc1[v]);
        }
    }
}

// ---------------------------------------------------------------- flash attention
// grid = (N/16, B), block = 128 (4 waves). Block owns 16 query rows.
// Key chunk = 128: wave w computes 2 score subtiles for keys [w*32, w*32+32),
// block-wide online softmax through LDS, then wave w accumulates out cols [w*64,+64).
__global__ __launch_bounds__(128) void gat_attn_kernel(
    const __bf16*  __restrict__ h1,   // [16384][256] queries
    const __bf16*  __restrict__ h2,   // [16384][256] keys
    const __bf16*  __restrict__ h3t,  // [8][256][2048] values (transposed)
    const unsigned* __restrict__ mb,  // [2048][64] packed adjacency
    float*         __restrict__ out)  // [8][2048][256]
{
    __shared__ __align__(16) __bf16 q_lds[16 * CDIM];   // 8 KB staged Q tile
    __shared__ __align__(16) __bf16 p_lds[16][128];     // 4 KB probs
    __shared__ float redmax[4][16];
    __shared__ float redsum[4][16];

    const int b     = blockIdx.y;
    const int q0    = blockIdx.x * 16;
    const int wave  = threadIdx.x >> 5;
    const int lane  = threadIdx.x & 31;
    const int lhalf = lane >> 4;
    const int l16   = lane & 15;
    const int row0  = 8 * lhalf;     // this lane's C-layout rows are v + row0

    // ---- stage Q tile (16x256 bf16 = 8KB) into LDS via async DMA, once per block
    {
        const __bf16* qg = h1 + (size_t)(b * NNODE + q0) * CDIM;
#pragma unroll
        for (int i = 0; i < 4; ++i) {
            int e = (threadIdx.x + i * 128) * 8;        // 8 bf16 = 16B per lane
            async_copy16(&q_lds[e], qg + e);
        }
        wait_async0();
        __syncthreads();
    }

    // ---- build Q A-fragments from LDS (K = 0..255 in 8 chunks)
    v16bf qf[8];
    {
        const __bf16* qrow = &q_lds[l16 * CDIM];
#pragma unroll
        for (int kc = 0; kc < 8; ++kc) {
            v8bf lo = *(const v8bf*)(qrow + kc * 32 + lhalf * 16);
            v8bf hi = *(const v8bf*)(qrow + kc * 32 + lhalf * 16 + 8);
            qf[kc] = __builtin_shufflevector(lo, hi,
                         0,1,2,3,4,5,6,7,8,9,10,11,12,13,14,15);
        }
    }

    v8f acc[4] = {v8f{}, v8f{}, v8f{}, v8f{}};
    float m_s[8], l_s[8];
#pragma unroll
    for (int v = 0; v < 8; ++v) { m_s[v] = -3.0e38f; l_s[v] = 0.0f; }

    for (int j = 0; j < NNODE / 128; ++j) {
        const int kchunk = j * 128;
        const int kb0    = kchunk + wave * 32;          // this wave's 32 keys

        // ---- two S subtiles = Q x K^T  (16 bf16 WMMAs over K=256)
        v8f s0 = {}, s1 = {};
#pragma unroll
        for (int kc = 0; kc < 8; ++kc) {
            v16bf kf0 = *(const v16bf*)(h2 + (size_t)(b * NNODE + kb0 + l16) * CDIM
                                           + kc * 32 + lhalf * 16);
            v16bf kf1 = *(const v16bf*)(h2 + (size_t)(b * NNODE + kb0 + 16 + l16) * CDIM
                                           + kc * 32 + lhalf * 16);
            s0 = __builtin_amdgcn_wmma_f32_16x16x32_bf16(false, qf[kc], false, kf0,
                                                         (short)0, s0, false, false);
            s1 = __builtin_amdgcn_wmma_f32_16x16x32_bf16(false, qf[kc], false, kf1,
                                                         (short)0, s1, false, false);
        }

        // ---- prefetch next chunk's K rows while we do softmax VALU work
        if (j + 1 < NNODE / 128) {
            const __bf16* nk = h2 + (size_t)(b * NNODE + kb0 + 128 + l16) * CDIM;
            __builtin_prefetch(nk, 0, 1);
            __builtin_prefetch(nk + 16 * CDIM, 0, 1);
        }

        // ---- leakyrelu + packed adjacency mask (1 u32 covers this wave's 32 keys)
#pragma unroll
        for (int v = 0; v < 8; ++v) {
            unsigned m = mb[(q0 + v + row0) * NWORD + (kb0 >> 5)];
            float x0 = s0[v]; x0 = x0 > 0.0f ? x0 : ALPHA * x0;
            float x1 = s1[v]; x1 = x1 > 0.0f ? x1 : ALPHA * x1;
            s0[v] = ((m >> l16) & 1u)        ? x0 : NEG_BIG;
            s1[v] = ((m >> (16 + l16)) & 1u) ? x1 : NEG_BIG;
        }

        // ---- row max over this wave's 32 keys (pairwise + 16-lane shfl tree)
        float rm[8];
#pragma unroll
        for (int v = 0; v < 8; ++v) {
            float x = fmaxf(s0[v], s1[v]);
            x = fmaxf(x, __shfl_xor(x, 1, 16));
            x = fmaxf(x, __shfl_xor(x, 2, 16));
            x = fmaxf(x, __shfl_xor(x, 4, 16));
            x = fmaxf(x, __shfl_xor(x, 8, 16));
            rm[v] = x;
        }
        if (l16 == 0)
#pragma unroll
            for (int v = 0; v < 8; ++v) redmax[wave][v + row0] = rm[v];
        __syncthreads();

        float mnew[8], scale[8];
#pragma unroll
        for (int v = 0; v < 8; ++v) {
            float cm = fmaxf(fmaxf(redmax[0][v + row0], redmax[1][v + row0]),
                             fmaxf(redmax[2][v + row0], redmax[3][v + row0]));
            float mn = fmaxf(m_s[v], cm);
            scale[v] = __expf(m_s[v] - mn);
            mnew[v]  = mn;
        }

        // ---- P = exp(S - m), row sums over both subtiles
        float rs[8];
#pragma unroll
        for (int v = 0; v < 8; ++v) {
            float p0 = __expf(s0[v] - mnew[v]);
            float p1 = __expf(s1[v] - mnew[v]);
            s0[v] = p0; s1[v] = p1;
            float x = p0 + p1;
            x += __shfl_xor(x, 1, 16);
            x += __shfl_xor(x, 2, 16);
            x += __shfl_xor(x, 4, 16);
            x += __shfl_xor(x, 8, 16);
            rs[v] = x;
        }
#pragma unroll
        for (int v = 0; v < 8; ++v) {
            p_lds[v + row0][wave * 32 + l16]      = f2bf(s0[v]);
            p_lds[v + row0][wave * 32 + 16 + l16] = f2bf(s1[v]);
        }
        if (l16 == 0)
#pragma unroll
            for (int v = 0; v < 8; ++v) redsum[wave][v + row0] = rs[v];
        __syncthreads();

        // ---- online update of l and rescale accumulators
#pragma unroll
        for (int v = 0; v < 8; ++v) {
            float cs = redsum[0][v + row0] + redsum[1][v + row0]
                     + redsum[2][v + row0] + redsum[3][v + row0];
            l_s[v] = l_s[v] * scale[v] + cs;
            m_s[v] = mnew[v];
        }
#pragma unroll
        for (int t = 0; t < 4; ++t)
#pragma unroll
            for (int v = 0; v < 8; ++v)
                acc[t][v] *= scale[v];

        // ---- P x V : this wave's 64 output columns, K=128 over 4 sub-chunks
#pragma unroll
        for (int kc = 0; kc < 4; ++kc) {
            v8bf plo = *(const v8bf*)(&p_lds[l16][kc * 32 + lhalf * 16]);
            v8bf phi = *(const v8bf*)(&p_lds[l16][kc * 32 + lhalf * 16 + 8]);
            v16bf pa = __builtin_shufflevector(plo, phi,
                           0,1,2,3,4,5,6,7,8,9,10,11,12,13,14,15);
#pragma unroll
            for (int t = 0; t < 4; ++t) {
                const __bf16* vp = h3t + (size_t)b * (CDIM * NNODE)
                                 + (wave * 64 + t * 16 + l16) * NNODE
                                 + kchunk + kc * 32 + lhalf * 16;
                v16bf vb = *(const v16bf*)vp;
                acc[t] = __builtin_amdgcn_wmma_f32_16x16x32_bf16(false, pa, false, vb,
                                                                 (short)0, acc[t],
                                                                 false, false);
            }
        }
        // barriers at top of next iteration protect p_lds/red* reuse
    }

    // ---- finalize: divide by l, ReLU, store f32
#pragma unroll
    for (int v = 0; v < 8; ++v) {
        float inv = 1.0f / l_s[v];
#pragma unroll
        for (int t = 0; t < 4; ++t) {
            float x = acc[t][v] * inv;
            x = x > 0.0f ? x : 0.0f;
            out[(size_t)(b * NNODE + q0 + v + row0) * CDIM + wave * 64 + t * 16 + l16] = x;
        }
    }
}

// ---------------------------------------------------------------- launch
extern "C" void kernel_launch(void* const* d_in, const int* in_sizes, int n_in,
                              void* d_out, int out_size, void* d_ws, size_t ws_size,
                              hipStream_t stream) {
    const float* inp = (const float*)d_in[0];
    const int*   adj = (const int*)d_in[1];
    const float* w1  = (const float*)d_in[2];
    const float* w2  = (const float*)d_in[3];
    const float* w3  = (const float*)d_in[4];
    float* out = (float*)d_out;

    char* ws = (char*)d_ws;
    const size_t XB_SZ = (size_t)NROW * CDIM * 2;         // 8 MB
    const size_t WT_SZ = (size_t)3 * CDIM * CDIM * 2;     // 384 KB
    const size_t H_SZ  = XB_SZ;                           // 8 MB each
    const size_t MB_OFF = XB_SZ + WT_SZ + 3 * H_SZ;
    __bf16*   xb  = (__bf16*)(ws);
    __bf16*   wt  = (__bf16*)(ws + XB_SZ);
    __bf16*   h1  = (__bf16*)(ws + XB_SZ + WT_SZ);
    __bf16*   h2  = (__bf16*)(ws + XB_SZ + WT_SZ + H_SZ);
    __bf16*   h3t = (__bf16*)(ws + XB_SZ + WT_SZ + 2 * H_SZ);
    unsigned* mb  = (unsigned*)(ws + MB_OFF);             // 512 KB

    // 1) convert inputs to bf16
    {
        int n = NROW * CDIM;
        cvt_x_kernel<<<(n + 255) / 256, 256, 0, stream>>>(inp, xb, n);
    }
    // 2) convert + transpose weights
    {
        int n = 3 * CDIM * CDIM;
        cvt_w_kernel<<<(n + 255) / 256, 256, 0, stream>>>(w1, w2, w3, wt);
    }
    // 3) pack adjacency into bitmasks
    {
        int n = NNODE * NWORD;
        pack_adj_kernel<<<(n + 255) / 256, 256, 0, stream>>>(adj, mb);
    }
    // 4) three projections via WMMA (H3 transposed)
    {
        dim3 grid(NROW / 16, 1, 3);
        gemm_proj_kernel<<<grid, 256, 0, stream>>>(xb, wt, h1, h2, h3t);
    }
    // 5) fused masked-softmax attention (flash style, 128-key chunks)
    {
        dim3 grid(NNODE / 16, BATCH);
        gat_attn_kernel<<<grid, 128, 0, stream>>>(h1, h2, h3t, mb, out);
    }
}
